// LSTMLayer_45226005627001
// MI455X (gfx1250) — matrix operand order repaired and checked by reading, not compiled, not run on hardware
//
#include <hip/hip_runtime.h>
#include <hip/hip_bf16.h>
#include <stdint.h>
#include <stddef.h>

// ---------------- problem constants ----------------
#define T_STEPS 512
#define BATCH   64
#define IN_DIM  1024
#define HID     1024
#define GATES   4096          // 4*HID
#define KTOT    2048          // IN_DIM + HID (fused A = [x_t | h])
#define NT_TILES 256          // GATES/16
#define KT_TILES 64           // KTOT/32
#define NWG      64           // persistent workgroups (4 m-tiles x 64 j-tiles / 4 waves)
#define BLK      128          // 4 waves per workgroup

typedef __attribute__((ext_vector_type(16))) __bf16       v16bf;
typedef __attribute__((ext_vector_type(8)))  float        v8f;
typedef __attribute__((ext_vector_type(4)))  unsigned int u32x4;

union FragAB { u32x4 u[2]; v16bf v; };   // 32B per lane = 16 bf16

// ---------------- workspace layout (bytes) ----------------
static constexpr size_t WS_WPACK = 0;                                          // 2048*4096*2 = 16 MB
static constexpr size_t WS_XBF   = WS_WPACK + (size_t)KTOT * GATES * 2;        // 64 MB
static constexpr size_t WS_HBUF  = WS_XBF   + (size_t)T_STEPS * BATCH * IN_DIM * 2; // 2x[64,1024] bf16
static constexpr size_t WS_BIAS  = WS_HBUF  + 2 * (size_t)BATCH * HID * 2;     // 4096 f32
static constexpr size_t WS_CNT   = WS_BIAS  + (size_t)GATES * 4;               // barrier counter
// total ~84.2 MB

// ---------------- fast activations (v_exp_f32 / v_rcp_f32) ----------------
__device__ __forceinline__ float sigmoid_f(float x) {
    return __builtin_amdgcn_rcpf(1.0f + __expf(-x));
}
__device__ __forceinline__ float tanh_f(float x) {
    // 2*sigmoid(2x)-1 ; saturates cleanly at +/-1
    return 2.0f * __builtin_amdgcn_rcpf(1.0f + __expf(-2.0f * x)) - 1.0f;
}

// ---------------- setup kernel 1: pack W into WMMA B-fragment layout ----------------
// B matrix Bmat[k, n], k in [0,2048), n in [0,4096):
//   k <  1024 -> W_ih[n, k]          (W_ih is [4096,1024] row-major)
//   k >= 1024 -> W_hh[n, k-1024]
// Fragment (nT,kT): 32 lanes x 16 bf16.  lanes 0-15: N=lane, K=kT*32+{0..7,16..23} ;
//                                        lanes 16-31: N=lane-16, K=kT*32+{8..15,24..31}
__global__ void pack_weights_kernel(const float* __restrict__ Wih,
                                    const float* __restrict__ Whh,
                                    __hip_bfloat16* __restrict__ wp) {
    int tid = blockIdx.x * blockDim.x + threadIdx.x;
    if (tid >= NT_TILES * KT_TILES * 32) return;
    int lane = tid & 31;
    int kT   = (tid >> 5) & (KT_TILES - 1);
    int nT   = tid >> 11;
    int n     = nT * 16 + (lane & 15);
    int kbase = kT * 32 + ((lane >> 4) * 8);
    __hip_bfloat16* dst = wp + (((size_t)(nT * KT_TILES + kT)) * 32 + lane) * 16;
#pragma unroll
    for (int q = 0; q < 16; ++q) {
        int k = kbase + (q < 8 ? q : q + 8);
        float w = (k < IN_DIM) ? Wih[(size_t)n * IN_DIM + k]
                               : Whh[(size_t)n * HID + (k - IN_DIM)];
        dst[q] = __float2bfloat16(w);
    }
}

// ---------------- setup kernel 2: x fp32 -> bf16 ----------------
__global__ void cvt_x_kernel(const float* __restrict__ x,
                             __hip_bfloat16* __restrict__ xb, int n) {
    int stride = gridDim.x * blockDim.x;
    for (int i = blockIdx.x * blockDim.x + threadIdx.x; i < n; i += stride)
        xb[i] = __float2bfloat16(x[i]);
}

// ---------------- setup kernel 3: h0 -> bf16 buffer 0, fused bias, counter=0 ----------------
__global__ void init_state_kernel(const float* __restrict__ h0,
                                  const float* __restrict__ bih,
                                  const float* __restrict__ bhh,
                                  __hip_bfloat16* __restrict__ hbuf,
                                  float* __restrict__ bias,
                                  unsigned int* __restrict__ cnt) {
    int i = blockIdx.x * blockDim.x + threadIdx.x;
    if (i < BATCH * HID) hbuf[i] = __float2bfloat16(h0[i]);
    if (i < GATES)       bias[i] = bih[i] + bhh[i];
    if (i == 0)          *cnt = 0u;
}

// ---------------- persistent LSTM kernel ----------------
// grid = 64 WGs x 128 threads.  wave (m = waveId) x (j = blockIdx.x) owns the
// 16x16 tile rows [m*16,m*16+16) x hidden cols [j*16,j*16+16) for ALL gates;
// cell state c for that tile lives in VGPRs for the whole 512-step run.
__global__ __launch_bounds__(BLK, 1)
void lstm_persistent_kernel(const float* __restrict__ c0,
                            const __hip_bfloat16* __restrict__ wp,
                            const __hip_bfloat16* __restrict__ xb,
                            __hip_bfloat16* __restrict__ hbuf,
                            const float* __restrict__ bias,
                            unsigned int* __restrict__ cnt,
                            float* __restrict__ out) {
    const int lane = threadIdx.x & 31;
    const int m    = threadIdx.x >> 5;     // 0..3  (batch tile)
    const int j    = blockIdx.x;           // 0..63 (hidden-column tile)
    const int colT = lane & 15;
    const int hsel = lane >> 4;            // 0/1 -> A K-half / C row-half
    const int aRow = m * 16 + colT;        // batch row this lane's A-fragment covers
    const int jcol = j * 16 + colT;        // hidden column for epilogue
    const int kofs = hsel * 8;             // A fragment per-lane K offset (elements)

    // per-gate B fragment base pointers (gate g occupies n-tiles g*64+j)
    const __hip_bfloat16* wpg[4];
#pragma unroll
    for (int g = 0; g < 4; ++g)
        wpg[g] = wp + (((size_t)((g * 64 + j) * KT_TILES)) * 32 + lane) * 16;

    // fused bias per lane, per gate (i,f,g,o)
    float bgate[4];
#pragma unroll
    for (int g = 0; g < 4; ++g) bgate[g] = bias[g * HID + jcol];

    // register-resident cell state: VGPR r holds row m*16 + r + hsel*8
    float creg[8];
#pragma unroll
    for (int r = 0; r < 8; ++r)
        creg[r] = c0[(size_t)(m * 16 + r + hsel * 8) * HID + jcol];

    for (int t = 0; t < T_STEPS; ++t) {
        const __hip_bfloat16* hr = hbuf + (size_t)(t & 1) * BATCH * HID;
        __hip_bfloat16*       hw = hbuf + (size_t)((t + 1) & 1) * BATCH * HID;
        const __hip_bfloat16* xa = xb + (size_t)t * BATCH * IN_DIM
                                      + (size_t)aRow * IN_DIM + kofs;
        const __hip_bfloat16* ha = hr + (size_t)aRow * HID + kofs;

        v8f acc0 = {}, acc1 = {}, acc2 = {}, acc3 = {};

        // ---- K = [0,1024): x_t part of A ----
        for (int kt = 0; kt < 32; ++kt) {
            FragAB a;
            a.u[0] = *(const u32x4*)(const void*)(xa + kt * 32);
            a.u[1] = *(const u32x4*)(const void*)(xa + kt * 32 + 16);
            __builtin_prefetch(wpg[0] + (size_t)(kt + 4) * 512, 0, 3);
            FragAB b0, b1, b2, b3;
            const __hip_bfloat16* p0 = wpg[0] + (size_t)kt * 512;
            const __hip_bfloat16* p1 = wpg[1] + (size_t)kt * 512;
            const __hip_bfloat16* p2 = wpg[2] + (size_t)kt * 512;
            const __hip_bfloat16* p3 = wpg[3] + (size_t)kt * 512;
            b0.u[0] = *(const u32x4*)(const void*)(p0);
            b0.u[1] = *(const u32x4*)(const void*)(p0 + 8);
            b1.u[0] = *(const u32x4*)(const void*)(p1);
            b1.u[1] = *(const u32x4*)(const void*)(p1 + 8);
            b2.u[0] = *(const u32x4*)(const void*)(p2);
            b2.u[1] = *(const u32x4*)(const void*)(p2 + 8);
            b3.u[0] = *(const u32x4*)(const void*)(p3);
            b3.u[1] = *(const u32x4*)(const void*)(p3 + 8);
            acc0 = __builtin_amdgcn_wmma_f32_16x16x32_bf16(false, a.v, false, b0.v, (short)0, acc0, false, false);
            acc1 = __builtin_amdgcn_wmma_f32_16x16x32_bf16(false, a.v, false, b1.v, (short)0, acc1, false, false);
            acc2 = __builtin_amdgcn_wmma_f32_16x16x32_bf16(false, a.v, false, b2.v, (short)0, acc2, false, false);
            acc3 = __builtin_amdgcn_wmma_f32_16x16x32_bf16(false, a.v, false, b3.v, (short)0, acc3, false, false);
        }
        // ---- K = [1024,2048): h_{t-1} part of A ----
        for (int kt = 0; kt < 32; ++kt) {
            FragAB a;
            a.u[0] = *(const u32x4*)(const void*)(ha + kt * 32);
            a.u[1] = *(const u32x4*)(const void*)(ha + kt * 32 + 16);
            __builtin_prefetch(wpg[0] + (size_t)(32 + kt + 4) * 512, 0, 3);
            FragAB b0, b1, b2, b3;
            const __hip_bfloat16* p0 = wpg[0] + (size_t)(32 + kt) * 512;
            const __hip_bfloat16* p1 = wpg[1] + (size_t)(32 + kt) * 512;
            const __hip_bfloat16* p2 = wpg[2] + (size_t)(32 + kt) * 512;
            const __hip_bfloat16* p3 = wpg[3] + (size_t)(32 + kt) * 512;
            b0.u[0] = *(const u32x4*)(const void*)(p0);
            b0.u[1] = *(const u32x4*)(const void*)(p0 + 8);
            b1.u[0] = *(const u32x4*)(const void*)(p1);
            b1.u[1] = *(const u32x4*)(const void*)(p1 + 8);
            b2.u[0] = *(const u32x4*)(const void*)(p2);
            b2.u[1] = *(const u32x4*)(const void*)(p2 + 8);
            b3.u[0] = *(const u32x4*)(const void*)(p3);
            b3.u[1] = *(const u32x4*)(const void*)(p3 + 8);
            acc0 = __builtin_amdgcn_wmma_f32_16x16x32_bf16(false, a.v, false, b0.v, (short)0, acc0, false, false);
            acc1 = __builtin_amdgcn_wmma_f32_16x16x32_bf16(false, a.v, false, b1.v, (short)0, acc1, false, false);
            acc2 = __builtin_amdgcn_wmma_f32_16x16x32_bf16(false, a.v, false, b2.v, (short)0, acc2, false, false);
            acc3 = __builtin_amdgcn_wmma_f32_16x16x32_bf16(false, a.v, false, b3.v, (short)0, acc3, false, false);
        }

        // ---- epilogue: gates -> c,h ; C layout: VGPR r, lanes<16 row=r, lanes>=16 row=r+8
#pragma unroll
        for (int r = 0; r < 8; ++r) {
            float ig = sigmoid_f(acc0[r] + bgate[0]);
            float fg = sigmoid_f(acc1[r] + bgate[1]);
            float gg = tanh_f   (acc2[r] + bgate[2]);
            float og = sigmoid_f(acc3[r] + bgate[3]);
            float c  = fg * creg[r] + ig * gg;
            creg[r]  = c;
            float h  = og * tanh_f(c);
            int row  = m * 16 + r + hsel * 8;
            out[(size_t)t * BATCH * HID + (size_t)row * HID + jcol] = h;
            hw[(size_t)row * HID + jcol] = __float2bfloat16(h);
            if (t == T_STEPS - 1) {
                out[(size_t)T_STEPS * BATCH * HID + (size_t)row * HID + jcol] = h;        // hT
                out[(size_t)T_STEPS * BATCH * HID + (size_t)BATCH * HID
                    + (size_t)row * HID + jcol] = c;                                      // cT
            }
        }

        // ---- grid-wide barrier before next step reads hw ----
        if (t != T_STEPS - 1) {
            __threadfence();
            __syncthreads();
            if (threadIdx.x == 0) {
                __hip_atomic_fetch_add(cnt, 1u, __ATOMIC_RELEASE, __HIP_MEMORY_SCOPE_AGENT);
                unsigned int target = (unsigned int)(t + 1) * (unsigned int)gridDim.x;
                while (__hip_atomic_load(cnt, __ATOMIC_ACQUIRE, __HIP_MEMORY_SCOPE_AGENT) < target)
                    __builtin_amdgcn_s_sleep(1);
            }
            __syncthreads();
        }
    }
}

// ---------------- host entry ----------------
extern "C" void kernel_launch(void* const* d_in, const int* in_sizes, int n_in,
                              void* d_out, int out_size, void* d_ws, size_t ws_size,
                              hipStream_t stream) {
    (void)in_sizes; (void)n_in; (void)out_size; (void)ws_size; // needs ~84.3 MB of ws
    const float* x   = (const float*)d_in[0];
    const float* h0  = (const float*)d_in[1];
    const float* c0  = (const float*)d_in[2];
    const float* Wih = (const float*)d_in[3];
    const float* Whh = (const float*)d_in[4];
    const float* bih = (const float*)d_in[5];
    const float* bhh = (const float*)d_in[6];
    float* out = (float*)d_out;

    char* ws = (char*)d_ws;
    __hip_bfloat16* wp   = (__hip_bfloat16*)(ws + WS_WPACK);
    __hip_bfloat16* xbf  = (__hip_bfloat16*)(ws + WS_XBF);
    __hip_bfloat16* hbuf = (__hip_bfloat16*)(ws + WS_HBUF);
    float*          bias = (float*)(ws + WS_BIAS);
    unsigned int*   cnt  = (unsigned int*)(ws + WS_CNT);

    // setup (parallel, one-shot each call; deterministic)
    pack_weights_kernel<<<(NT_TILES * KT_TILES * 32 + 255) / 256, 256, 0, stream>>>(Wih, Whh, wp);
    cvt_x_kernel<<<4096, 256, 0, stream>>>(x, xbf, T_STEPS * BATCH * IN_DIM);
    init_state_kernel<<<(BATCH * HID + 255) / 256, 256, 0, stream>>>(h0, bih, bhh, hbuf, bias, cnt);

    // persistent recurrent kernel: 64 WGs x 128 threads (256 waves = 256 output tiles)
    lstm_persistent_kernel<<<NWG, BLK, 0, stream>>>(c0, wp, xbf, hbuf, bias, cnt, out);
}
